// WindowAttention_19181323944374
// MI455X (gfx1250) — compile-verified, matrix operand order checked by reading, not run on hardware
//
#include <hip/hip_runtime.h>
#include <hip/hip_bf16.h>

// ---------------------------------------------------------------------------
// Types for CDNA5 WMMA (wave32): D(f32 16x16) = A(bf16 16x32) * B(bf16 32x16) + C
// ---------------------------------------------------------------------------
typedef __bf16 bf16;
typedef __attribute__((ext_vector_type(16))) __bf16 v16bf;
typedef __attribute__((ext_vector_type(8)))  __bf16 v8bf;
typedef __attribute__((ext_vector_type(8)))  float  v8f;

struct __align__(8) us4 { unsigned short x, y, z, w; };

#define NUM_HEADS 12
#define HD 32
#define CDIM 384
#define LTOK 49
#define LPAD 64
#define NWIN 64
#define BWIN 2048
#define MROWS (BWIN * LTOK)          // 100352 tokens
#define QKV_N (3 * CDIM)             // 1152
#define KSTEPS (CDIM / 32)           // 12
#define SCALE_Q 0.17677669529663687f // 32^-0.5

// ---------------------------------------------------------------------------
// bf16 <-> f32 helpers (round-to-nearest-even)
// ---------------------------------------------------------------------------
__device__ __forceinline__ unsigned short f2u_bf(float f) {
  union { float f; unsigned u; } v; v.f = f;
  unsigned r = v.u + 0x7FFFu + ((v.u >> 16) & 1u);
  return (unsigned short)(r >> 16);
}
__device__ __forceinline__ bf16 u2bf(unsigned short s) {
  union { unsigned short s; bf16 b; } o; o.s = s; return o.b;
}
__device__ __forceinline__ bf16 f2bf(float f) { return u2bf(f2u_bf(f)); }
__device__ __forceinline__ float bf2f(bf16 b) {
  union { bf16 b; unsigned short s; } i; i.b = b;
  union { unsigned u; float f; } o; o.u = ((unsigned)i.s) << 16;
  return o.f;
}

// ---------------------------------------------------------------------------
// CDNA5 async copy: 16 bytes/lane, global -> LDS, tracked by ASYNCcnt.
// lds_off = low 32 bits of generic address of an LDS location (== LDS byte
// offset per ISA flat-aperture rules); ga = 64-bit global address.
// ---------------------------------------------------------------------------
__device__ __forceinline__ void async_copy16(void* lds_ptr, const void* gptr) {
  unsigned lds_off = (unsigned)(size_t)lds_ptr;
  unsigned long long ga = (unsigned long long)(size_t)gptr;
  asm volatile("global_load_async_to_lds_b128 %0, %1, off"
               :: "v"(lds_off), "v"(ga) : "memory");
}
__device__ __forceinline__ void wait_async0() {
  asm volatile("s_wait_asynccnt 0x0" ::: "memory");
}
__device__ __forceinline__ void wait_async3() {
  asm volatile("s_wait_asynccnt 0x3" ::: "memory");
}

// ---------------------------------------------------------------------------
// WMMA wrapper + fragment load.
// ISA 16-bit A/B layout: lanes 0-15 hold K={0..7,16..23}, lanes 16-31
// K={8..15,24..31}; kbase = (lane<16 ? 0 : 8). Two 16B LDS loads.
// ---------------------------------------------------------------------------
__device__ __forceinline__ v8f wmma_bf16(v16bf a, v16bf b, v8f c) {
  return __builtin_amdgcn_wmma_f32_16x16x32_bf16(
      false, a, false, b, (short)0, c, false, false);
}
__device__ __forceinline__ v16bf load_frag(const bf16* row, int kofs) {
  v8bf lo = *(const v8bf*)(row + kofs);
  v8bf hi = *(const v8bf*)(row + kofs + 16);
  v16bf r;
#pragma unroll
  for (int e = 0; e < 8; ++e) { r[e] = lo[e]; r[e + 8] = hi[e]; }
  return r;
}

// ---------------------------------------------------------------------------
// Prep kernel: f32 -> bf16 streaming convert (x), vectorized 4/thread.
// ---------------------------------------------------------------------------
__global__ void convert_x(const float* __restrict__ in,
                          unsigned short* __restrict__ out, int n4) {
  int i = (blockIdx.x * 256 + threadIdx.x);
  if (i >= n4) return;
  float4 v = *(const float4*)(in + (size_t)i * 4);
  us4 o;
  o.x = f2u_bf(v.x); o.y = f2u_bf(v.y); o.z = f2u_bf(v.z); o.w = f2u_bf(v.w);
  *(us4*)(out + (size_t)i * 4) = o;
}

// Prep kernel: weight [K][N] f32 -> [N][K] bf16 (so B-tile rows are contiguous
// along K and can be async-copied straight into LDS).
__global__ void convert_wT(const float* __restrict__ in,
                           unsigned short* __restrict__ out, int K, int N) {
  int gid = blockIdx.x * 256 + threadIdx.x;
  if (gid >= K * N) return;
  int n = gid / K, k = gid % K;
  out[(size_t)n * K + k] = f2u_bf(in[(size_t)k * N + n]);
}

// ---------------------------------------------------------------------------
// RPE gather: layout [i][h][j][c] in f32 (L2-resident; pure v_fmac consumers).
// rpe_table row reshaped [H][3*hd]: q=+0, k=+32, v=+64 within a head's 96.
// ---------------------------------------------------------------------------
__global__ void rpe_prep(const float* __restrict__ tbl,
                         float* __restrict__ qr,
                         float* __restrict__ kr,
                         float* __restrict__ vr) {
  int gid = blockIdx.x * blockDim.x + threadIdx.x;
  const int total = LTOK * LTOK * NUM_HEADS * HD;
  if (gid >= total) return;
  int c = gid & (HD - 1);
  int t = gid >> 5;
  int h = t % NUM_HEADS; t /= NUM_HEADS;
  int j = t % LTOK;
  int i = t / LTOK;
  int ih = i / 7, iw = i % 7, jh = j / 7, jw = j % 7;
  int ridx = (ih - jh + 6) * 13 + (iw - jw + 6);
  const float* row = tbl + (size_t)ridx * QKV_N;
  size_t o = ((size_t)((i * NUM_HEADS + h) * LTOK + j)) * HD + c;
  int col = h * (3 * HD) + c;
  qr[o] = row[col] * SCALE_Q;          // q_rpe pre-scaled
  kr[o] = row[col + HD];
  vr[o] = row[col + 2 * HD];
}

// ---------------------------------------------------------------------------
// Kernel: QKV GEMM  [100352 x 384]bf16 * [384 x 1152]bf16 + bias -> bf16.
// Double-buffered async-to-LDS tiles; 3 async instrs/wave/tile.
// Block tile 64(M) x 128(N); 8 waves = 4(M)x2(N), 4 accumulators each.
// ---------------------------------------------------------------------------
__global__ void qkv_gemm(const unsigned short* __restrict__ xbf,
                         const unsigned short* __restrict__ wT,  // [1152][384]
                         const float* __restrict__ bias,
                         unsigned short* __restrict__ out) {
  __shared__ bf16 As[2][64][32];    // [m][k]
  __shared__ bf16 Bs[2][128][32];   // [n][k]
  const int m0 = blockIdx.x * 64;
  const int n0 = blockIdx.y * 128;
  const int tid = threadIdx.x;
  const int lane = tid & 31;
  const int wid = tid >> 5;
  const int wm = wid & 3;
  const int wn = wid >> 2;
  const int kbase = (lane < 16) ? 0 : 8;
  const int lr = lane & 15;

  v8f acc[4];
#pragma unroll
  for (int t = 0; t < 4; ++t) acc[t] = (v8f){};

  // per-thread tile-chunk assignments (16B chunks)
  const int ar = tid >> 2, ap = (tid & 3) * 8;          // A: 256 chunks
  auto issue = [&](int buf, int k0) {
    async_copy16(&As[buf][ar][ap], xbf + (size_t)(m0 + ar) * CDIM + k0 + ap);
#pragma unroll
    for (int it = 0; it < 2; ++it) {                    // B: 512 chunks
      int ch = tid * 2 + it;
      int n = ch >> 2, p = (ch & 3) * 8;
      async_copy16(&Bs[buf][n][p], wT + (size_t)(n0 + n) * CDIM + k0 + p);
    }
  };

  issue(0, 0);
  for (int kt = 0; kt < KSTEPS; ++kt) {
    int buf = kt & 1;
    if (kt + 1 < KSTEPS) { issue(buf ^ 1, (kt + 1) * 32); wait_async3(); }
    else                 { wait_async0(); }
    __syncthreads();
    v16bf af = load_frag(&As[buf][wm * 16 + lr][0], kbase);
#pragma unroll
    for (int t = 0; t < 4; ++t) {
      v16bf bfrag = load_frag(&Bs[buf][wn * 64 + t * 16 + lr][0], kbase);
      acc[t] = wmma_bf16(af, bfrag, acc[t]);
    }
    __syncthreads();
  }
#pragma unroll
  for (int t = 0; t < 4; ++t) {
#pragma unroll
    for (int r = 0; r < 8; ++r) {
      int m = m0 + wm * 16 + r + ((lane < 16) ? 0 : 8);
      int n = n0 + wn * 64 + t * 16 + lr;
      float v = acc[t][r] + bias[n];
      if (n < CDIM) v *= SCALE_Q;              // fold q scaling
      out[(size_t)m * QKV_N + n] = f2u_bf(v);
    }
  }
}

// ---------------------------------------------------------------------------
// Fused window attention, one block per (window b, head h).
// ---------------------------------------------------------------------------
__global__ void attn_kernel(const unsigned short* __restrict__ qkv,
                            const float* __restrict__ mask,
                            const float* __restrict__ qrpe,
                            const float* __restrict__ krpe,
                            const float* __restrict__ vrpe,
                            unsigned short* __restrict__ out_heads) {
  const int b = blockIdx.x, h = blockIdx.y;
  const int tid = threadIdx.x, lane = tid & 31, wid = tid >> 5;
  const int kbase = (lane < 16) ? 0 : 8;
  const int lr = lane & 15;

  __shared__ bf16  qs[LPAD][HD];     // bf16 q (pre-scaled) for WMMA
  __shared__ bf16  ks[LPAD][HD];
  __shared__ bf16  vt[HD][LPAD];     // v transposed for P.V B-frags
  __shared__ float qf[LPAD][HD];     // f32 shadows for VALU RPE dots
  __shared__ float kf[LPAD][HD];
  __shared__ float S[LPAD][68];      // logits
  __shared__ bf16  P[LPAD][LPAD];    // probs (bf16, WMMA A)
  __shared__ float Pf[LPAD][LPAD];   // probs (f32, VALU)
  __shared__ float O[LPAD][36];      // P.V accumulators

  const size_t tokbase = (size_t)b * LTOK;

  // zero-pad rows 49..63 of qs/ks (disjoint from async targets)
  for (int idx = tid; idx < (LPAD - LTOK) * HD * 2; idx += 256) {
    int m = idx >= (LPAD - LTOK) * HD;
    int r = LTOK + (idx % ((LPAD - LTOK) * HD)) / HD;
    int c = idx & 31;
    (m ? ks : qs)[r][c] = u2bf(0);
  }
  // v: sync transposed load
#pragma unroll
  for (int it = 0; it < 8; ++it) {
    int idx = tid * 8 + it;
    int r = idx >> 5, c = idx & 31;
    unsigned short vv = 0;
    if (r < LTOK) vv = qkv[(tokbase + r) * QKV_N + 2 * CDIM + h * HD + c];
    vt[c][r] = u2bf(vv);
  }
  // q,k: async 16B chunks straight into LDS (49 rows x 4 chunks x 2 mats)
  for (int ch = tid; ch < 2 * LTOK * 4; ch += 256) {
    int isk = ch >= LTOK * 4;
    int cc = isk ? ch - LTOK * 4 : ch;
    int r = cc >> 2, p = (cc & 3) * 8;
    const unsigned short* g =
        qkv + (tokbase + r) * QKV_N + (isk ? CDIM : 0) + h * HD + p;
    async_copy16(isk ? &ks[r][p] : &qs[r][p], g);
  }
  wait_async0();
  __syncthreads();

  // f32 shadows of q,k for the VALU RPE dot products
#pragma unroll
  for (int it = 0; it < 8; ++it) {
    int idx = tid * 8 + it;
    int r = idx >> 5, c = idx & 31;
    qf[r][c] = bf2f(qs[r][c]);
    kf[r][c] = bf2f(ks[r][c]);
  }

  // ---- S = Q K^T : 4x4 tiles of 16x16 (K=32=hd, single WMMA each); 2/wave
#pragma unroll
  for (int tt = 0; tt < 2; ++tt) {
    int tile = wid * 2 + tt;
    int ti = tile >> 2, tj = tile & 3;
    v16bf af = load_frag(&qs[ti * 16 + lr][0], kbase);
    v16bf bfrag = load_frag(&ks[tj * 16 + lr][0], kbase);  // B = K^T
    v8f acc = (v8f){};
    acc = wmma_bf16(af, bfrag, acc);
#pragma unroll
    for (int r = 0; r < 8; ++r)
      S[ti * 16 + r + ((lane < 16) ? 0 : 8)][tj * 16 + lr] = acc[r];
  }
  __syncthreads();

  // ---- RPE terms + mask + softmax. 4 lanes per row, 16 j each.
  {
    int i = tid >> 2;
    int s = tid & 3;
    float vals[16];
    float mx = -3.0e38f;
#pragma unroll
    for (int u = 0; u < 16; ++u) {
      int j = s * 16 + u;
      float v = -3.0e38f;
      if (i < LTOK && j < LTOK) {
        size_t rbase = ((size_t)((i * NUM_HEADS + h) * LTOK + j)) * HD;
        const float* krp = krpe + rbase;
        const float* qrp = qrpe + rbase;
        float a1 = 0.f, a2 = 0.f;
#pragma unroll
        for (int c = 0; c < HD; ++c) {
          a1 = __builtin_fmaf(qf[i][c], krp[c], a1);  // (q*scale) . k_rpe
          a2 = __builtin_fmaf(kf[j][c], qrp[c], a2);  // k . (q_rpe*scale)
        }
        v = S[i][j] + a1 + a2 +
            mask[((size_t)(b & (NWIN - 1)) * LTOK + i) * LTOK + j];
      }
      vals[u] = v;
      mx = fmaxf(mx, v);
    }
    mx = fmaxf(mx, __shfl_xor(mx, 1, 32));
    mx = fmaxf(mx, __shfl_xor(mx, 2, 32));
    float sum = 0.f;
#pragma unroll
    for (int u = 0; u < 16; ++u) {
      float e = (vals[u] > -1.0e38f) ? __expf(vals[u] - mx) : 0.f;
      vals[u] = e;
      sum += e;
    }
    sum += __shfl_xor(sum, 1, 32);
    sum += __shfl_xor(sum, 2, 32);
    float inv = (i < LTOK) ? (1.f / sum) : 0.f;
#pragma unroll
    for (int u = 0; u < 16; ++u) {
      float p = vals[u] * inv;
      P[i][s * 16 + u] = f2bf(p);
      Pf[i][s * 16 + u] = p;
    }
  }
  __syncthreads();

  // ---- O = P V : 4(M) x 2(N) tiles, one per wave, two K-steps of 32.
  {
    int ti = wid & 3, tj = wid >> 2;
    v8f acc = (v8f){};
#pragma unroll
    for (int kk = 0; kk < 2; ++kk) {
      v16bf af = load_frag(&P[ti * 16 + lr][kk * 32], kbase);
      v16bf bfrag = load_frag(&vt[tj * 16 + lr][kk * 32], kbase);
      acc = wmma_bf16(af, bfrag, acc);
    }
#pragma unroll
    for (int r = 0; r < 8; ++r)
      O[ti * 16 + r + ((lane < 16) ? 0 : 8)][tj * 16 + lr] = acc[r];
  }
  __syncthreads();

  // ---- + P . v_rpe, write bf16 [token][C] for the projection GEMM.
  for (int idx = tid; idx < LTOK * HD; idx += 256) {
    int i = idx >> 5, c = idx & 31;
    float acc = O[i][c];
    const float* vrp = vrpe + ((size_t)(i * NUM_HEADS + h) * LTOK) * HD + c;
#pragma unroll 7
    for (int j = 0; j < LTOK; ++j)
      acc = __builtin_fmaf(Pf[i][j], vrp[(size_t)j * HD], acc);
    out_heads[(tokbase + i) * CDIM + h * HD + c] = f2u_bf(acc);
  }
}

// ---------------------------------------------------------------------------
// Projection GEMM  [100352 x 384]bf16 * [384 x 384]bf16 + bias -> f32 out.
// Same double-buffered async structure as qkv_gemm.
// ---------------------------------------------------------------------------
__global__ void proj_gemm(const unsigned short* __restrict__ a,
                          const unsigned short* __restrict__ wT, // [384][384]
                          const float* __restrict__ bias,
                          float* __restrict__ out) {
  __shared__ bf16 As[2][64][32];
  __shared__ bf16 Bs[2][128][32];
  const int m0 = blockIdx.x * 64;
  const int n0 = blockIdx.y * 128;
  const int tid = threadIdx.x;
  const int lane = tid & 31;
  const int wid = tid >> 5;
  const int wm = wid & 3;
  const int wn = wid >> 2;
  const int kbase = (lane < 16) ? 0 : 8;
  const int lr = lane & 15;

  v8f acc[4];
#pragma unroll
  for (int t = 0; t < 4; ++t) acc[t] = (v8f){};

  const int ar = tid >> 2, ap = (tid & 3) * 8;
  auto issue = [&](int buf, int k0) {
    async_copy16(&As[buf][ar][ap], a + (size_t)(m0 + ar) * CDIM + k0 + ap);
#pragma unroll
    for (int it = 0; it < 2; ++it) {
      int ch = tid * 2 + it;
      int n = ch >> 2, p = (ch & 3) * 8;
      async_copy16(&Bs[buf][n][p], wT + (size_t)(n0 + n) * CDIM + k0 + p);
    }
  };

  issue(0, 0);
  for (int kt = 0; kt < KSTEPS; ++kt) {
    int buf = kt & 1;
    if (kt + 1 < KSTEPS) { issue(buf ^ 1, (kt + 1) * 32); wait_async3(); }
    else                 { wait_async0(); }
    __syncthreads();
    v16bf af = load_frag(&As[buf][wm * 16 + lr][0], kbase);
#pragma unroll
    for (int t = 0; t < 4; ++t) {
      v16bf bfrag = load_frag(&Bs[buf][wn * 64 + t * 16 + lr][0], kbase);
      acc[t] = wmma_bf16(af, bfrag, acc[t]);
    }
    __syncthreads();
  }
#pragma unroll
  for (int t = 0; t < 4; ++t) {
#pragma unroll
    for (int r = 0; r < 8; ++r) {
      int m = m0 + wm * 16 + r + ((lane < 16) ? 0 : 8);
      int n = n0 + wn * 64 + t * 16 + lr;
      out[(size_t)m * CDIM + n] = acc[t][r] + bias[n];
    }
  }
}

// ---------------------------------------------------------------------------
// Host launcher
// ---------------------------------------------------------------------------
extern "C" void kernel_launch(void* const* d_in, const int* in_sizes, int n_in,
                              void* d_out, int out_size, void* d_ws, size_t ws_size,
                              hipStream_t stream) {
  const float* x         = (const float*)d_in[0];
  const float* attn_mask = (const float*)d_in[1];
  const float* qkv_w     = (const float*)d_in[2];
  const float* qkv_b     = (const float*)d_in[3];
  const float* rpe_table = (const float*)d_in[4];
  const float* proj_w    = (const float*)d_in[5];
  const float* proj_b    = (const float*)d_in[6];
  float* out = (float*)d_out;

  char* ws = (char*)d_ws;
  size_t off = 0;
  unsigned short* qkv_bf = (unsigned short*)(ws + off);
  off += (size_t)MROWS * QKV_N * 2;                       // 231 MB
  unsigned short* out_heads = (unsigned short*)(ws + off);
  off += (size_t)MROWS * CDIM * 2;                        // 77 MB
  unsigned short* x_bf = (unsigned short*)(ws + off);
  off += (size_t)MROWS * CDIM * 2;                        // 77 MB
  unsigned short* wqkvT = (unsigned short*)(ws + off);
  off += (size_t)QKV_N * CDIM * 2;
  unsigned short* wprojT = (unsigned short*)(ws + off);
  off += (size_t)CDIM * CDIM * 2;
  const size_t rpe_elems = (size_t)LTOK * LTOK * NUM_HEADS * HD;
  float* qrpe = (float*)(ws + off); off += rpe_elems * 4;
  float* krpe = (float*)(ws + off); off += rpe_elems * 4;
  float* vrpe = (float*)(ws + off); off += rpe_elems * 4;

  const int xn4 = MROWS * CDIM / 4;
  convert_x<<<(xn4 + 255) / 256, 256, 0, stream>>>(x, x_bf, xn4);
  convert_wT<<<(CDIM * QKV_N + 255) / 256, 256, 0, stream>>>(qkv_w, wqkvT, CDIM, QKV_N);
  convert_wT<<<(CDIM * CDIM + 255) / 256, 256, 0, stream>>>(proj_w, wprojT, CDIM, CDIM);
  rpe_prep<<<(int)((rpe_elems + 255) / 256), 256, 0, stream>>>(
      rpe_table, qrpe, krpe, vrpe);

  qkv_gemm<<<dim3(MROWS / 64, QKV_N / 128), 256, 0, stream>>>(
      x_bf, wqkvT, qkv_b, qkv_bf);

  attn_kernel<<<dim3(BWIN, NUM_HEADS), 256, 0, stream>>>(
      qkv_bf, attn_mask, qrpe, krpe, vrpe, out_heads);

  proj_gemm<<<dim3(MROWS / 64, CDIM / 128), 256, 0, stream>>>(
      out_heads, wprojT, proj_b, out);
}
